// QuatNetPytorchDenseLayer_68221260529948
// MI455X (gfx1250) — matrix operand order repaired and checked by reading, not compiled, not run on hardware
//
#include <hip/hip_runtime.h>
#include <hip/hip_bf16.h>

// QuatNet dense layer as a 4096x2048x2048 f16-WMMA GEMM + bias + tanh.
// BATCH=4096, IN_Q=512, OUT_Q=512 -> X[4096][2048], A[2048][2048] (Hamilton-expanded W)

typedef __attribute__((ext_vector_type(16))) _Float16 v16h;
typedef __attribute__((ext_vector_type(8)))  _Float16 v8h;
typedef __attribute__((ext_vector_type(4)))  _Float16 v4h;
typedef __attribute__((ext_vector_type(8)))  float    v8f;

#define KDIM 2048   // 4*IN_Q
#define NDIM 2048   // 4*OUT_Q
#define MDIM 4096   // BATCH

#if __has_builtin(__builtin_amdgcn_global_load_async_to_lds_b128)
#define HAVE_ASYNC_LDS 1
#else
#define HAVE_ASYNC_LDS 0
#endif

// b128 async payload type, matching the builtin's parameter:
//   '__attribute__((__vector_size__(4 * sizeof(int)))) int *'
typedef int v4i __attribute__((vector_size(4 * sizeof(int))));
typedef __attribute__((address_space(1))) v4i* gas1_v4i;
typedef __attribute__((address_space(3))) v4i* lds3_v4i;

__device__ __forceinline__ void wait_async_zero() {
#if __has_builtin(__builtin_amdgcn_s_wait_asynccnt)
  __builtin_amdgcn_s_wait_asynccnt(0);
#else
  asm volatile("s_wait_asynccnt 0x0" ::: "memory");
#endif
}

// Branch-free tanh: 1 - 2/(e^{2v}+1), via v_exp_f32 + v_rcp_f32.
__device__ __forceinline__ float fast_tanh(float v) {
  const float e = __builtin_amdgcn_exp2f(v * 2.885390082f);  // 2*log2(e)
  return 1.0f - 2.0f * __builtin_amdgcn_rcpf(e + 1.0f);
}

// ---------------------------------------------------------------------------
// Expand quaternion weights W[o][i][4] (f32) into f16 matrix Wt[od][ic],
// od = o*4+d (row, "N"-major so B-tiles load contiguously), ic = i*4+c.
// ---------------------------------------------------------------------------
__global__ void quat_expand_w(const float* __restrict__ W, _Float16* __restrict__ Wt) {
  const int t = blockIdx.x * blockDim.x + threadIdx.x;   // 0 .. 512*512-1
  const int o = t >> 9;
  const int i = t & 511;
  const float4 q = *reinterpret_cast<const float4*>(W + (size_t)t * 4);
  const float w = q.x, x = q.y, y = q.z, z = q.w;
  _Float16* base = Wt + ((size_t)o * 4) * KDIM + (size_t)i * 4;
  const v4h r0 = { (_Float16)  w, (_Float16)(-x), (_Float16)(-y), (_Float16)(-z) };
  const v4h r1 = { (_Float16)  x, (_Float16)  w,  (_Float16)(-z), (_Float16)  y  };
  const v4h r2 = { (_Float16)  y, (_Float16)  z,  (_Float16)  w,  (_Float16)(-x) };
  const v4h r3 = { (_Float16)  z, (_Float16)(-y), (_Float16)  x,  (_Float16)  w  };
  *reinterpret_cast<v4h*>(base + 0 * KDIM) = r0;
  *reinterpret_cast<v4h*>(base + 1 * KDIM) = r1;
  *reinterpret_cast<v4h*>(base + 2 * KDIM) = r2;
  *reinterpret_cast<v4h*>(base + 3 * KDIM) = r3;
}

// ---------------------------------------------------------------------------
// Convert X f32 -> f16 once (so GEMM staging is pure b128 async copies, and
// X re-reads across the 16 N-blocks move half the bytes).
// ---------------------------------------------------------------------------
__global__ void conv_x_f16(const float* __restrict__ X, _Float16* __restrict__ Xh) {
  const size_t t = (size_t)blockIdx.x * blockDim.x + threadIdx.x;  // per float4
  const float4 f = *reinterpret_cast<const float4*>(X + t * 4);
  const v4h h = { (_Float16)f.x, (_Float16)f.y, (_Float16)f.z, (_Float16)f.w };
  *reinterpret_cast<v4h*>(Xh + t * 4) = h;
}

// ---------------------------------------------------------------------------
// GEMM: Out[4096][2048] = tanh( Xh * Wt^T + bias ), f32 out.
// 256 threads (8 waves), tile 128(M) x 128(N), BK=32, double-buffered LDS,
// GLOBAL_LOAD_ASYNC_TO_LDS_B128 staging, v_wmma_f32_16x16x32_f16 compute.
// ---------------------------------------------------------------------------
#define LDS_PITCH 40   // 32 + 8 halves padding -> conflict-free b128 LDS loads

__global__ __launch_bounds__(256)
void quat_gemm_tanh(const _Float16* __restrict__ Xh, const _Float16* __restrict__ Wt,
                    const float* __restrict__ Bias, float* __restrict__ Out) {
  __shared__ _Float16 As[2][128][LDS_PITCH];  // [m][k]
  __shared__ _Float16 Bs[2][128][LDS_PITCH];  // [n][k]

  const int tid   = threadIdx.x;
  const int lane  = tid & 31;
  const int wid   = tid >> 5;
  const int waveM = wid >> 2;        // 0..1
  const int waveN = wid & 3;         // 0..3
  const int g     = lane >> 4;       // half-wave group (selects K sub-range)
  const int ln    = lane & 15;
  const int mBase = blockIdx.y * 128;
  const int nBase = blockIdx.x * 128;

  // Stage one BK=32 slab (A: 128x32 halves, B: 128x32 halves) into buffer buf.
  // 512 b128 transfers per matrix over 256 threads = 2 each.
  auto stage = [&](int buf, int k0) {
    #pragma unroll
    for (int j = 0; j < 2; ++j) {
      const int idx = tid + j * 256;        // 0..511
      const int r   = idx >> 2;             // row 0..127
      const int c   = (idx & 3) << 3;       // k col 0,8,16,24
      const _Float16* gA = Xh + (size_t)(mBase + r) * KDIM + (size_t)(k0 + c);
      const _Float16* gB = Wt + (size_t)(nBase + r) * KDIM + (size_t)(k0 + c);
#if HAVE_ASYNC_LDS
      __builtin_amdgcn_global_load_async_to_lds_b128(
          (gas1_v4i)(void*)gA, (lds3_v4i)(void*)&As[buf][r][c], 0, 0);
      __builtin_amdgcn_global_load_async_to_lds_b128(
          (gas1_v4i)(void*)gB, (lds3_v4i)(void*)&Bs[buf][r][c], 0, 0);
#else
      *reinterpret_cast<v8h*>(&As[buf][r][c]) = *reinterpret_cast<const v8h*>(gA);
      *reinterpret_cast<v8h*>(&Bs[buf][r][c]) = *reinterpret_cast<const v8h*>(gB);
#endif
    }
  };

  v8f acc[4][2];
  const v8f vzero = {0.f, 0.f, 0.f, 0.f, 0.f, 0.f, 0.f, 0.f};
  #pragma unroll
  for (int mt = 0; mt < 4; ++mt)
    #pragma unroll
    for (int nt = 0; nt < 2; ++nt)
      acc[mt][nt] = vzero;

  stage(0, 0);
#if HAVE_ASYNC_LDS
  wait_async_zero();
#endif
  __syncthreads();

  const int KT = KDIM / 32;   // 64 K-steps
  for (int kt = 0; kt < KT; ++kt) {
    const int cur = kt & 1;
    if (kt + 1 < KT) stage(cur ^ 1, (kt + 1) * 32);  // async prefetch next slab

    // A fragments: 16x32 f16 tile (ISA 7.12.2):
    //   lanes 0-15: halves 0-7 -> K=0..7,  halves 8-15 -> K=16..23
    //   lanes16-31: halves 0-7 -> K=8..15, halves 8-15 -> K=24..31
    v16h a[4];
    #pragma unroll
    for (int mt = 0; mt < 4; ++mt) {
      const int row = waveM * 64 + mt * 16 + ln;
      const v8h lo = *reinterpret_cast<const v8h*>(&As[cur][row][8 * g]);
      const v8h hi = *reinterpret_cast<const v8h*>(&As[cur][row][16 + 8 * g]);
      a[mt] = __builtin_shufflevector(lo, hi, 0,1,2,3,4,5,6,7,8,9,10,11,12,13,14,15);
    }
    // B fragments: 32x16 tile, lane = column N, same K order as A (halves 0-7 -> K=8*g+h, halves 8-15 -> K=16+8*g+(h-8)).
    v16h b[2];
    #pragma unroll
    for (int nt = 0; nt < 2; ++nt) {
      const int row = waveN * 32 + nt * 16 + ln;
      const v8h lo = *reinterpret_cast<const v8h*>(&Bs[cur][row][8 * g]);
      const v8h hi = *reinterpret_cast<const v8h*>(&Bs[cur][row][16 + 8 * g]);
      b[nt] = __builtin_shufflevector(lo, hi, 0,1,2,3,4,5,6,7,8,9,10,11,12,13,14,15);
    }

    #pragma unroll
    for (int mt = 0; mt < 4; ++mt)
      #pragma unroll
      for (int nt = 0; nt < 2; ++nt)
        acc[mt][nt] = __builtin_amdgcn_wmma_f32_16x16x32_f16(
            /*neg_a=*/false, a[mt], /*neg_b=*/false, b[nt],
            /*c_mod=*/(short)0, acc[mt][nt],
            /*reuse_a=*/false, /*reuse_b=*/false);

#if HAVE_ASYNC_LDS
    wait_async_zero();   // prefetched slab resident before next iteration reads it
#endif
    __syncthreads();
  }

  // Epilogue: C/D layout -> VGPR r: lanes 0-15 M=r, lanes 16-31 M=8+r; N=lane&15.
  #pragma unroll
  for (int nt = 0; nt < 2; ++nt) {
    const int col = nBase + waveN * 32 + nt * 16 + ln;
    const float bv = Bias[col];
    #pragma unroll
    for (int mt = 0; mt < 4; ++mt) {
      const int row0 = mBase + waveM * 64 + mt * 16 + g * 8;
      #pragma unroll
      for (int r = 0; r < 8; ++r) {
        Out[(size_t)(row0 + r) * NDIM + col] = fast_tanh(acc[mt][nt][r] + bv);
      }
    }
  }
}

// ---------------------------------------------------------------------------
extern "C" void kernel_launch(void* const* d_in, const int* in_sizes, int n_in,
                              void* d_out, int out_size, void* d_ws, size_t ws_size,
                              hipStream_t stream) {
  (void)in_sizes; (void)n_in; (void)out_size; (void)ws_size;
  const float* x  = (const float*)d_in[0];   // [4096][512][4]
  const float* W  = (const float*)d_in[1];   // [512][512][4]
  const float* B  = (const float*)d_in[2];   // [512][4] -> flat [2048]
  float* out = (float*)d_out;                // [4096][2048]

  _Float16* wt = (_Float16*)d_ws;            // 2048*2048 f16 = 8 MiB
  _Float16* xh = wt + (size_t)NDIM * KDIM;   // 4096*2048 f16 = 16 MiB

  quat_expand_w<<<(512 * 512) / 256, 256, 0, stream>>>(W, wt);
  conv_x_f16<<<(MDIM * KDIM / 4) / 256, 256, 0, stream>>>(x, xh);

  dim3 grid(NDIM / 128, MDIM / 128);         // (16, 32)
  quat_gemm_tanh<<<grid, 256, 0, stream>>>(xh, wt, B, out);
}
